// ScaledDotProductAttention_11879879542874
// MI455X (gfx1250) — matrix operand order repaired and checked by reading, not compiled, not run on hardware
//
#include <hip/hip_runtime.h>
#include <hip/hip_bf16.h>
#include <math.h>

typedef __bf16 v16bf __attribute__((ext_vector_type(16)));
typedef __bf16 v8bf  __attribute__((ext_vector_type(8)));
typedef __bf16 v2bf  __attribute__((ext_vector_type(2)));
typedef float  v8f   __attribute__((ext_vector_type(8)));
typedef float  v4f   __attribute__((ext_vector_type(4)));

#define BATCH 4
#define HEADS 16
#define SQL   2048
#define SKL   2048
#define DH    64
#define NEGC  (-1.0e9f)
#define SCALE 0.125f                    // 1/sqrt(64)
#define LOG2E 1.4426950408889634f
#define S2    (SCALE * LOG2E)           // fold scale + log2 conversion into one fma
#define NEGL  (NEGC * LOG2E)
#define WAVES 4
#define VSTR  40                        // padded col stride (elems) of transposed V in LDS

// compile-time XOR lane shuffle via ds_swizzle_b32 (group-of-32 mode)
template <int IMM>
__device__ __forceinline__ float swz_xor(float x) {
    return __builtin_bit_cast(float,
        __builtin_amdgcn_ds_swizzle(__builtin_bit_cast(int, x), IMM));
}

// QK^T 16x16 logits tile: 2x v_wmma_f32_16x16x32_bf16 over d=0..63.
__device__ __forceinline__ v8f qk_tile(const float* __restrict__ kr,
                                       const v16bf& a0, const v16bf& a1)
{
    v16bf b0, b1;
#pragma unroll
    for (int e = 0; e < 16; ++e) { b0[e] = (__bf16)kr[e]; b1[e] = (__bf16)kr[32 + e]; }
    v8f c = {};
    c = __builtin_amdgcn_wmma_f32_16x16x32_bf16(false, a0, false, b0, (short)0, c, false, false);
    c = __builtin_amdgcn_wmma_f32_16x16x32_bf16(false, a1, false, b1, (short)0, c, false, false);
    return c;
}

__global__ __launch_bounds__(WAVES * 32)
void fa_fwd_wmma(const float* __restrict__ q, const float* __restrict__ k,
                 const float* __restrict__ v, const float* __restrict__ mask,
                 float* __restrict__ out, float* __restrict__ wts)
{
    __shared__ __bf16 ldsP[WAVES][16 * 32];   // per-wave P staging (16 rows x 32 keys)
    __shared__ __bf16 ldsV[64 * VSTR];        // block-shared transposed V chunk [col][row]

    const int tid  = threadIdx.x;
    const int wave = tid >> 5;
    const int lane = tid & 31;
    const int n    = lane & 15;     // column within 16-wide tile (B/C/D layouts)
    const int h    = lane >> 4;     // half-wave id
    const int h8   = h * 8;

    const int qblks = SQL / (16 * WAVES);
    const int bh    = blockIdx.x / qblks;
    const int qb    = blockIdx.x % qblks;
    const int b     = bh / HEADS;
    const int q0    = qb * (16 * WAVES) + wave * 16;   // this wave's 16-query tile
    const int qmaxb = qb * (16 * WAVES) + 16 * WAVES - 1;  // block's last query row

    const float* __restrict__ Qp = q + (size_t)bh * SQL * DH + (size_t)(q0 + n) * DH;
    const float* __restrict__ Kp = k + (size_t)bh * SKL * DH;
    const float* __restrict__ Vp = v + (size_t)bh * SKL * DH;
    const float* __restrict__ Mp = mask + (size_t)b * SKL;

    // ---- Q A-fragments (16x32 bf16 each): d 0..31 and d 32..63. Row m == n for A layout.
    v16bf a0, a1;
#pragma unroll
    for (int e = 0; e < 8; ++e) {
        a0[e]     = (__bf16)Qp[h * 16 + e];
        a0[e + 8] = (__bf16)Qp[h * 16 + 8 + e];
        a1[e]     = (__bf16)Qp[32 + h * 16 + e];
        a1[e + 8] = (__bf16)Qp[32 + h * 16 + 8 + e];
    }

    const int ktmax = q0 >> 4;      // diagonal 16-key tile index

    // ================= PASS 1a: per-row max (log2 domain), max-only =================
    float rmax[8];
#pragma unroll
    for (int r = 0; r < 8; ++r) rmax[r] = -3.0e38f;

    for (int kt = 0; kt <= ktmax; ++kt) {
        const int jg = kt * 16 + n;
        v8f c = qk_tile(Kp + (size_t)jg * DH + h * 16, a0, a1);
        const float mt = Mp[jg] * NEGL;
        if (kt == ktmax) {          // wave-uniform: only tile with causal masking
#pragma unroll
            for (int r = 0; r < 8; ++r) {
                float x = fmaf(c[r], S2, mt) + ((jg > q0 + h8 + r) ? NEGL : 0.0f);
                rmax[r] = fmaxf(rmax[r], x);
            }
        } else {
#pragma unroll
            for (int r = 0; r < 8; ++r)
                rmax[r] = fmaxf(rmax[r], fmaf(c[r], S2, mt));
        }
    }
#pragma unroll
    for (int r = 0; r < 8; ++r) {
        rmax[r] = fmaxf(rmax[r], swz_xor<0x041f>(rmax[r]));   // SWAPX1
        rmax[r] = fmaxf(rmax[r], swz_xor<0x081f>(rmax[r]));   // SWAPX2
        rmax[r] = fmaxf(rmax[r], swz_xor<0x101f>(rmax[r]));   // SWAPX4
        rmax[r] = fmaxf(rmax[r], swz_xor<0x201f>(rmax[r]));   // SWAPX8
    }

    // ================= PASS 1b: row sum of exp2(x - rowmax), raw v_exp_f32 =========
    float rsum[8];
#pragma unroll
    for (int r = 0; r < 8; ++r) rsum[r] = 0.0f;

    for (int kt = 0; kt <= ktmax; ++kt) {
        const int jg = kt * 16 + n;
        v8f c = qk_tile(Kp + (size_t)jg * DH + h * 16, a0, a1);
        const float mt = Mp[jg] * NEGL;
        if (kt == ktmax) {
#pragma unroll
            for (int r = 0; r < 8; ++r) {
                float x = fmaf(c[r], S2, mt) + ((jg > q0 + h8 + r) ? NEGL : 0.0f);
                rsum[r] += __builtin_amdgcn_exp2f(x - rmax[r]);
            }
        } else {
#pragma unroll
            for (int r = 0; r < 8; ++r)
                rsum[r] += __builtin_amdgcn_exp2f(fmaf(c[r], S2, mt - rmax[r]));
        }
    }
#pragma unroll
    for (int r = 0; r < 8; ++r) {
        rsum[r] += swz_xor<0x041f>(rsum[r]);
        rsum[r] += swz_xor<0x081f>(rsum[r]);
        rsum[r] += swz_xor<0x101f>(rsum[r]);
        rsum[r] += swz_xor<0x201f>(rsum[r]);
    }
    float rinv[8];
#pragma unroll
    for (int r = 0; r < 8; ++r) rinv[r] = __builtin_amdgcn_rcpf(rsum[r]);

    // ================= PASS 2: write normalized P, accumulate O = P @ V =============
    v8f o0 = {}, o1 = {}, o2 = {}, o3 = {};
    __bf16* myP = &ldsP[wave][0];
    float* __restrict__ wbase = wts + ((size_t)bh * SQL + q0) * SKL;

    // V staging thread mapping: row-pair rp = tid&15 (rows 2rp,2rp+1), col group g = tid>>4
    const int rp = tid & 15;
    const int g  = tid >> 4;

    for (int ktp = 0; ktp < SKL / 32; ++ktp) {
        const int  kbase     = ktp * 32;
        const bool blocklive = (kbase <= qmaxb);   // block-uniform
        const bool live      = (kbase <= q0 + 15); // wave-uniform; live => blocklive

        // ---- (1) weights for two 16-col tiles + wave-local P staging
#pragma unroll
        for (int t = 0; t < 2; ++t) {
            const int kcol0 = kbase + t * 16;
            const int jg    = kcol0 + n;
            float p[8];
            if (kcol0 <= q0 + 15) {                // wave-uniform
                const int kt = kcol0 >> 4;
                v8f c = qk_tile(Kp + (size_t)jg * DH + h * 16, a0, a1);
                const float mt = Mp[jg] * NEGL;
                if (kt == ktmax) {
#pragma unroll
                    for (int r = 0; r < 8; ++r) {
                        float x = fmaf(c[r], S2, mt) + ((jg > q0 + h8 + r) ? NEGL : 0.0f);
                        p[r] = __builtin_amdgcn_exp2f(x - rmax[r]) * rinv[r];
                    }
                } else {
#pragma unroll
                    for (int r = 0; r < 8; ++r)
                        p[r] = __builtin_amdgcn_exp2f(fmaf(c[r], S2, mt - rmax[r])) * rinv[r];
                }
            } else {
#pragma unroll
                for (int r = 0; r < 8; ++r) p[r] = 0.0f;
            }
            // weights fully written (harness poisons d_out), incl. zero triangle
#pragma unroll
            for (int r = 0; r < 8; ++r)
                __builtin_nontemporal_store(p[r], wbase + (size_t)(h8 + r) * SKL + jg);
            if (live) {
#pragma unroll
                for (int r = 0; r < 8; ++r)
                    myP[(h8 + r) * 32 + t * 16 + n] = (__bf16)p[r];
            }
        }

        // ---- (2) cooperative transposed V staging (block-uniform barriers)
        if (blocklive) {
            __syncthreads();   // prior iteration's ldsV reads are complete
            const float* vr0 = Vp + (size_t)(kbase + 2 * rp) * DH + g * 8;
            v4f r0a = *(const v4f*)vr0,        r0b = *(const v4f*)(vr0 + 4);
            v4f r1a = *(const v4f*)(vr0 + DH), r1b = *(const v4f*)(vr0 + DH + 4);
#pragma unroll
            for (int i = 0; i < 8; ++i) {
                float e0 = (i < 4) ? r0a[i] : r0b[i - 4];
                float e1 = (i < 4) ? r1a[i] : r1b[i - 4];
                v2bf pk; pk[0] = (__bf16)e0; pk[1] = (__bf16)e1;   // rows 2rp, 2rp+1
                *(v2bf*)(&ldsV[(g * 8 + i) * VSTR + 2 * rp]) = pk;
            }
            __syncthreads();   // staged chunk visible to all waves
        }

        // ---- (3) P @ V for this 32-key chunk
        if (live) {
            asm volatile("s_wait_dscnt 0x0" ::: "memory");   // wave-local P writes done
            v8bf plo = *(const v8bf*)(myP + n * 32 + h * 16);
            v8bf phi = *(const v8bf*)(myP + n * 32 + h * 16 + 8);
            v16bf aP;
#pragma unroll
            for (int e = 0; e < 8; ++e) { aP[e] = plo[e]; aP[e + 8] = phi[e]; }
            asm volatile("" ::: "memory");
            // B-fragments straight out of transposed LDS: 2x ds_load_b128 per chunk
            v8f* oacc[4] = { &o0, &o1, &o2, &o3 };
#pragma unroll
            for (int dc = 0; dc < 4; ++dc) {
                const __bf16* vc = &ldsV[(dc * 16 + n) * VSTR + h * 16];
                v8bf x0 = *(const v8bf*)vc;
                v8bf x1 = *(const v8bf*)(vc + 8);
                v16bf bv;
#pragma unroll
                for (int e = 0; e < 8; ++e) { bv[e] = x0[e]; bv[e + 8] = x1[e]; }
                *oacc[dc] = __builtin_amdgcn_wmma_f32_16x16x32_bf16(
                    false, aP, false, bv, (short)0, *oacc[dc], false, false);
            }
        }
    }

    // ---- write O tile (16 x 64)
    float* __restrict__ obase = out + ((size_t)bh * SQL + q0) * DH;
#pragma unroll
    for (int r = 0; r < 8; ++r) {
        float* orow = obase + (size_t)(h8 + r) * DH + n;
        __builtin_nontemporal_store(o0[r], orow);
        __builtin_nontemporal_store(o1[r], orow + 16);
        __builtin_nontemporal_store(o2[r], orow + 32);
        __builtin_nontemporal_store(o3[r], orow + 48);
    }
}

extern "C" void kernel_launch(void* const* d_in, const int* in_sizes, int n_in,
                              void* d_out, int out_size, void* d_ws, size_t ws_size,
                              hipStream_t stream) {
    const float* q    = (const float*)d_in[0];
    const float* k    = (const float*)d_in[1];
    const float* v    = (const float*)d_in[2];
    const float* mask = (const float*)d_in[3];
    float* out = (float*)d_out;                                // output first...
    float* wts = out + (size_t)BATCH * HEADS * SQL * DH;       // ...then attention weights

    dim3 grid(BATCH * HEADS * (SQL / (16 * WAVES)));
    dim3 block(WAVES * 32);
    hipLaunchKernelGGL(fa_fwd_wmma, grid, block, 0, stream, q, k, v, mask, out, wts);
}
